// TinyformerEncoder_84095459655823
// MI455X (gfx1250) — matrix operand order repaired and checked by reading, not compiled, not run on hardware
//
#include <hip/hip_runtime.h>
#include <hip/hip_bf16.h>
#include <math.h>

// ---------------------------------------------------------------------------
// Tinyformer encoder on MI455X (gfx1250). Compute-bound (~470 GFLOP vs <1 GB
// HBM traffic at 23.3 TB/s) -> every matmul via v_wmma_f32_16x16x32_bf16.
// All tile shapes divide the problem exactly: NO bounds checks anywhere.
// LDS layouts are fragment-contiguous so gathers lower to ds_load_b128.
// ---------------------------------------------------------------------------

typedef __attribute__((ext_vector_type(16))) __bf16 v16bf;
typedef __attribute__((ext_vector_type(8)))  __bf16 v8bf;
typedef __attribute__((ext_vector_type(4)))  __bf16 v4bf;
typedef __attribute__((ext_vector_type(8)))  float  v8f;
typedef __attribute__((ext_vector_type(4)))  float  v4f;

#define D_MODEL 512
#define D_ATTN  128
#define D_FF    2048
#define NB      8
#define SEQ     2048
#define ROWS    (NB * SEQ)   // 16384
#define NLAYER  4

// ---------------------------------------------------------------------------
// Fragment gathers (CDNA5 ISA 7.12.2). Layouts chosen so each lane's 16
// values are two runs of 8 contiguous bf16 in LDS -> two ds_load_b128 per fragment.
// ---------------------------------------------------------------------------

// A 16x32 from row-major [m][k] tile: lane half -> kbase 0/8; per lane the
// fragment is 8 contiguous bf16 at k=kb and 8 at k=16+kb.
__device__ __forceinline__ v16bf frag_A_lds(const __bf16* base, int stride,
                                            int row0, int lane) {
  int m  = row0 + (lane & 15);
  int kb = (lane >> 4) * 8;
  const __bf16* p = base + m * stride + kb;
  union { v16bf v; v8bf o[2]; } u;
  u.o[0] = *reinterpret_cast<const v8bf*>(p);
  u.o[1] = *reinterpret_cast<const v8bf*>(p + 16);
  return u.v;
}

// B 32x16 from TRANSPOSED [n][k] storage: lane half -> kbase 0/8; per lane
// the fragment is 8 contiguous bf16 at k=kb and 8 at k=16+kb.
__device__ __forceinline__ v16bf frag_B_ldsT(const __bf16* base, int stride,
                                             int col0, int koff, int lane) {
  int n  = col0 + (lane & 15);
  int kb = (lane >> 4) * 8;
  const __bf16* p = base + n * stride + koff + kb;
  union { v16bf v; v8bf o[2]; } u;
  u.o[0] = *reinterpret_cast<const v8bf*>(p);
  u.o[1] = *reinterpret_cast<const v8bf*>(p + 16);
  return u.v;
}

__device__ __forceinline__ v8f vzero8() {
  v8f z = {0.f, 0.f, 0.f, 0.f, 0.f, 0.f, 0.f, 0.f};
  return z;
}

__device__ __forceinline__ v4bf cvt4(v4f f) {
  v4bf r;
#pragma unroll
  for (int j = 0; j < 4; ++j) r[j] = (__bf16)f[j];
  return r;
}

// ---------------------------------------------------------------------------
// Generic WMMA GEMM: C[M,N] = sum_b A_b[M,K] @ op(B_b)[K,N] (+bias)(+res)(+relu)
// BM=128 BN=128 BK=32, 256 threads = 8 waves (4x2), 32x64 per wave
// (2x4 fragments -> 8 WMMAs per k-step). TRANSB compile-time.
// Requires M%128==0, N%128==0, K%32==0 (true for all call sites).
// ---------------------------------------------------------------------------
template <int TRANSB>
__global__ __launch_bounds__(256) void wmma_gemm(
    const float* __restrict__ A, int lda, long long sA,
    const float* __restrict__ B, int ldb, long long sB,
    const float* __restrict__ bias,
    const float* __restrict__ res,
    float* __restrict__ C, int ldc,
    int M, int N, int K, int nbatch, int relu)
{
  __shared__ __bf16 As[128][32];    // [m][k]
  __shared__ __bf16 BsT[128][32];   // [n][k]  (fragment-contiguous)

  const int tid  = threadIdx.x;
  const int lane = tid & 31;
  const int wave = tid >> 5;
  const int wr   = wave >> 1;       // 0..3  -> rows 32*wr
  const int wc   = wave & 1;        // 0..1  -> cols 64*wc
  const int m0   = blockIdx.y * 128;
  const int n0   = blockIdx.x * 128;

  // A staging decomposition: thread -> (row, 16-element half-row)
  const int ar = tid >> 1;
  const int ac = (tid & 1) << 4;

  v8f acc[2][4];
#pragma unroll
  for (int i = 0; i < 2; ++i)
#pragma unroll
    for (int j = 0; j < 4; ++j) acc[i][j] = vzero8();

  for (int b = 0; b < nbatch; ++b) {
    const float* Ab = A + (long long)b * sA;
    const float* Bb = B + (long long)b * sB;
    for (int k0 = 0; k0 < K; k0 += 32) {
      __syncthreads();
      // ---- stage A (128x32): 16 contiguous floats/thread, 4x b128 loads ----
      {
        const float* gp = Ab + (long long)(m0 + ar) * lda + k0 + ac;
        v4f f0 = *reinterpret_cast<const v4f*>(gp);
        v4f f1 = *reinterpret_cast<const v4f*>(gp + 4);
        v4f f2 = *reinterpret_cast<const v4f*>(gp + 8);
        v4f f3 = *reinterpret_cast<const v4f*>(gp + 12);
        *reinterpret_cast<v4bf*>(&As[ar][ac])      = cvt4(f0);
        *reinterpret_cast<v4bf*>(&As[ar][ac + 4])  = cvt4(f1);
        *reinterpret_cast<v4bf*>(&As[ar][ac + 8])  = cvt4(f2);
        *reinterpret_cast<v4bf*>(&As[ar][ac + 12]) = cvt4(f3);
      }
      // ---- stage B (32x128 logical) into BsT[n][k] ----
      if (TRANSB == 0) {
        // global [K,N]: float4 over n, scatter into BsT columns
        const int bn = (tid & 31) << 2;      // 0..124
        const int bk = tid >> 5;             // 0..7
#pragma unroll
        for (int kk = 0; kk < 4; ++kk) {
          int k = bk + kk * 8;
          v4f f = *reinterpret_cast<const v4f*>(
              Bb + (long long)(k0 + k) * ldb + n0 + bn);
          v4bf h = cvt4(f);
#pragma unroll
          for (int j = 0; j < 4; ++j) BsT[bn + j][k] = h[j];
        }
      } else {
        // global [N,K]: float4 over k, contiguous b64 stores into BsT
        const int bk = (tid & 7) << 2;       // 0..28
        const int bn = tid >> 3;             // 0..31
#pragma unroll
        for (int nn = 0; nn < 4; ++nn) {
          int n = bn + nn * 32;
          v4f f = *reinterpret_cast<const v4f*>(
              Bb + (long long)(n0 + n) * ldb + k0 + bk);
          *reinterpret_cast<v4bf*>(&BsT[n][bk]) = cvt4(f);
        }
      }
      // prefetch next A K-tile (global_prefetch_b8)
      if (k0 + 32 < K)
        __builtin_prefetch(Ab + (long long)(m0 + ar) * lda + k0 + 32 + ac, 0, 3);
      __syncthreads();

      v16bf af[2], bfr[4];
#pragma unroll
      for (int mi = 0; mi < 2; ++mi)
        af[mi] = frag_A_lds(&As[0][0], 32, wr * 32 + mi * 16, lane);
#pragma unroll
      for (int ni = 0; ni < 4; ++ni)
        bfr[ni] = frag_B_ldsT(&BsT[0][0], 32, wc * 64 + ni * 16, 0, lane);
#pragma unroll
      for (int mi = 0; mi < 2; ++mi)
#pragma unroll
        for (int ni = 0; ni < 4; ++ni)
          acc[mi][ni] = __builtin_amdgcn_wmma_f32_16x16x32_bf16(
              false, af[mi], false, bfr[ni], (short)0, acc[mi][ni], false, false);
    }
  }

  // ---- epilogue (C layout: element e -> row e + (lane>=16 ? 8:0)) ----
  const int halfoff = (lane >> 4) << 3;
  float bv[4] = {0.f, 0.f, 0.f, 0.f};
  if (bias) {
#pragma unroll
    for (int ni = 0; ni < 4; ++ni)
      bv[ni] = bias[n0 + wc * 64 + ni * 16 + (lane & 15)];
  }
#pragma unroll
  for (int mi = 0; mi < 2; ++mi)
#pragma unroll
    for (int ni = 0; ni < 4; ++ni) {
      int rbase = m0 + wr * 32 + mi * 16 + halfoff;
      int gc    = n0 + wc * 64 + ni * 16 + (lane & 15);
#pragma unroll
      for (int e = 0; e < 8; ++e) {
        long long gi = (long long)(rbase + e) * ldc + gc;
        float v = acc[mi][ni][e] + bv[ni];
        if (res)  v += res[gi];
        if (relu) v = fmaxf(v, 0.f);
        C[gi] = v;
      }
    }
}

// ---------------------------------------------------------------------------
// LayerNorm: one wave per row of 512, 8 rows per 256-thread block.
// ---------------------------------------------------------------------------
__global__ __launch_bounds__(256) void ln_kernel(
    const float* __restrict__ x, const float* __restrict__ w,
    const float* __restrict__ b, float* __restrict__ y,
    float eps)
{
  const int lane = threadIdx.x & 31;
  const int wave = threadIdx.x >> 5;
  long long row = (long long)blockIdx.x * 8 + wave;
  const float* xr = x + row * D_MODEL;
  float v[16];
  float s = 0.f;
#pragma unroll
  for (int i = 0; i < 16; ++i) { v[i] = xr[lane + i * 32]; s += v[i]; }
#pragma unroll
  for (int off = 1; off < 32; off <<= 1) s += __shfl_xor(s, off, 32);
  float mu = s * (1.0f / D_MODEL);
  float q = 0.f;
#pragma unroll
  for (int i = 0; i < 16; ++i) { float d = v[i] - mu; q += d * d; }
#pragma unroll
  for (int off = 1; off < 32; off <<= 1) q += __shfl_xor(q, off, 32);
  float rstd = rsqrtf(q * (1.0f / D_MODEL) + eps);
  float* yr = y + row * D_MODEL;
#pragma unroll
  for (int i = 0; i < 16; ++i) {
    int c = lane + i * 32;
    yr[c] = (v[i] - mu) * rstd * w[c] + b[c];
  }
}

// ---------------------------------------------------------------------------
// Flash attention: grid (S/64, B), 128 threads = 4 waves x 16 query rows.
// Keys in chunks of 32: QK^T (8 WMMA) -> +pos bias, causal, scale -> online
// softmax (shfl_xor half-wave reductions, v_exp_f32) -> P via LDS relayout ->
// PV (8 WMMA).
// ---------------------------------------------------------------------------
__global__ __launch_bounds__(128) void flash_attn(
    const float* __restrict__ qkv,      // [B,S,384] (q|k|v)
    const float* __restrict__ posA,     // [S,S]
    const float* __restrict__ kmask,    // [B,S]
    float* __restrict__ out,            // [B,S,128]
    int S)
{
  __shared__ __bf16 Kt[32][128];        // [key][d] : B-frag contiguous (k=d)
  __shared__ __bf16 Vt[128][32];        // [d][key] : B-frag contiguous (k=key)
  __shared__ __bf16 Ps[4][16][32];      // per-wave P staging

  const int tid  = threadIdx.x;
  const int lane = tid & 31;
  const int wave = tid >> 5;
  const int b    = blockIdx.y;
  const int q0   = blockIdx.x * 64 + wave * 16;
  const long long base = (long long)b * S * 384;
  const float scale = 0.0625f;          // 1/sqrt(2*D_ATTN)

  // Q A-fragments (d chunks of 32), vectorized: 8-float contiguous runs
  v16bf qf[4];
  {
    int m  = q0 + (lane & 15);
    int kb = (lane >> 4) * 8;
    const float* qp = qkv + base + (long long)m * 384 + kb;
#pragma unroll
    for (int dc = 0; dc < 4; ++dc) {
      const float* p0 = qp + dc * 32;
      const float* p1 = qp + dc * 32 + 16;
      v4f a0 = *reinterpret_cast<const v4f*>(p0);
      v4f a1 = *reinterpret_cast<const v4f*>(p0 + 4);
      v4f b0 = *reinterpret_cast<const v4f*>(p1);
      v4f b1 = *reinterpret_cast<const v4f*>(p1 + 4);
      union { v16bf v; v4bf q[4]; } u;
      u.q[0] = cvt4(a0); u.q[1] = cvt4(a1);
      u.q[2] = cvt4(b0); u.q[3] = cvt4(b1);
      qf[dc] = u.v;
    }
  }

  v8f acc[8];
#pragma unroll
  for (int dc = 0; dc < 8; ++dc) acc[dc] = vzero8();
  float mr[8], lr[8];
#pragma unroll
  for (int e = 0; e < 8; ++e) { mr[e] = -__builtin_inff(); lr[e] = 0.f; }

  const int halfoff = (lane >> 4) << 3;
  const int nchunks = (blockIdx.x + 1) * 2;     // causal bound

  for (int kc = 0; kc < nchunks; ++kc) {
    const int key0 = kc * 32;
    __syncthreads();
    // ---- cooperative K/V stage: float4 over d, 8 iterations/thread ----
#pragma unroll
    for (int it = 0; it < 8; ++it) {
      int idx = (tid + it * 128) << 2;         // element index, d-major
      int key = idx >> 7;
      int d   = idx & 127;
      const float* rp = qkv + base + (long long)(key0 + key) * 384;
      v4f kv = *reinterpret_cast<const v4f*>(rp + 128 + d);
      v4f vv = *reinterpret_cast<const v4f*>(rp + 256 + d);
      *reinterpret_cast<v4bf*>(&Kt[key][d]) = cvt4(kv);
      v4bf vh = cvt4(vv);
#pragma unroll
      for (int j = 0; j < 4; ++j) Vt[d + j][key] = vh[j];
    }
    __syncthreads();

    // ---- scores S = Q @ K^T ----
    v8f sc[2];
#pragma unroll
    for (int ni = 0; ni < 2; ++ni) {
      sc[ni] = vzero8();
#pragma unroll
      for (int dc = 0; dc < 4; ++dc) {
        v16bf kf = frag_B_ldsT(&Kt[0][0], 128, ni * 16, dc * 32, lane);
        sc[ni] = __builtin_amdgcn_wmma_f32_16x16x32_bf16(
            false, qf[dc], false, kf, (short)0, sc[ni], false, false);
      }
    }

    // ---- +pos bias, causal mask, scale ----
#pragma unroll
    for (int ni = 0; ni < 2; ++ni) {
      int key = key0 + ni * 16 + (lane & 15);
#pragma unroll
      for (int e = 0; e < 8; ++e) {
        int qr = q0 + halfoff + e;
        sc[ni][e] = (key <= qr)
                        ? (sc[ni][e] + posA[(long long)qr * S + key]) * scale
                        : -__builtin_inff();
      }
    }

    // ---- online softmax (row stats replicated across 16-lane halves) ----
#pragma unroll
    for (int e = 0; e < 8; ++e) {
      float t = fmaxf(sc[0][e], sc[1][e]);
#pragma unroll
      for (int off = 1; off < 16; off <<= 1) t = fmaxf(t, __shfl_xor(t, off, 32));
      float newm  = fmaxf(mr[e], t);
      float alpha = __expf(mr[e] - newm);
      float p0 = __expf(sc[0][e] - newm);
      float p1 = __expf(sc[1][e] - newm);
      float rs = p0 + p1;
#pragma unroll
      for (int off = 1; off < 16; off <<= 1) rs += __shfl_xor(rs, off, 32);
      lr[e] = lr[e] * alpha + rs;
      mr[e] = newm;
      sc[0][e] = p0; sc[1][e] = p1;
#pragma unroll
      for (int dc = 0; dc < 8; ++dc) acc[dc][e] *= alpha;
    }

    // ---- stage P (C-layout) -> reload as A-fragment ----
#pragma unroll
    for (int ni = 0; ni < 2; ++ni)
#pragma unroll
      for (int e = 0; e < 8; ++e)
        Ps[wave][halfoff + e][ni * 16 + (lane & 15)] = (__bf16)sc[ni][e];
    asm volatile("s_wait_dscnt 0" ::: "memory");  // same-wave LDS RAW fence
    v16bf pf = frag_A_lds(&Ps[wave][0][0], 32, 0, lane);

    // ---- acc += P @ V ----
#pragma unroll
    for (int dc = 0; dc < 8; ++dc) {
      v16bf vf = frag_B_ldsT(&Vt[0][0], 32, dc * 16, 0, lane);
      acc[dc] = __builtin_amdgcn_wmma_f32_16x16x32_bf16(
          false, pf, false, vf, (short)0, acc[dc], false, false);
    }
  }

  // ---- epilogue: o = (acc / l) * key_mask ----
#pragma unroll
  for (int e = 0; e < 8; ++e) {
    int qr = q0 + halfoff + e;
    float inv = 1.0f / lr[e];
    float km  = kmask[(long long)b * S + qr];
#pragma unroll
    for (int dc = 0; dc < 8; ++dc) {
      int d = dc * 16 + (lane & 15);
      out[(((long long)b * S + qr) << 7) + d] = acc[dc][e] * inv * km;
    }
  }
}

// ---------------------------------------------------------------------------
// Host orchestration
// ---------------------------------------------------------------------------
static inline void launch_gemm(hipStream_t st,
    const float* A, int lda, long long sA,
    const float* B, int ldb, long long sB, int transB,
    const float* bias, const float* res,
    float* C, int ldc, int M, int N, int K, int nb, int relu)
{
  dim3 g((unsigned)(N / 128), (unsigned)(M / 128));
  if (transB)
    wmma_gemm<1><<<g, 256, 0, st>>>(A, lda, sA, B, ldb, sB, bias, res, C, ldc,
                                    M, N, K, nb, relu);
  else
    wmma_gemm<0><<<g, 256, 0, st>>>(A, lda, sA, B, ldb, sB, bias, res, C, ldc,
                                    M, N, K, nb, relu);
}

extern "C" void kernel_launch(void* const* d_in, const int* in_sizes, int n_in,
                              void* d_out, int out_size, void* d_ws, size_t ws_size,
                              hipStream_t stream) {
  (void)in_sizes; (void)n_in; (void)out_size; (void)ws_size;
  const float* x_in      = (const float*)d_in[0];
  const float* pos_emb   = (const float*)d_in[1];
  const float* key_mask  = (const float*)d_in[2];
  const float* pos_qk_W  = (const float*)d_in[3];
  const float* pos_qk_b  = (const float*)d_in[4];
  const float* qkv_W     = (const float*)d_in[5];
  const float* qkv_b     = (const float*)d_in[6];
  const float* out_W     = (const float*)d_in[7];
  const float* out_b     = (const float*)d_in[8];
  const float* attn_ln_w = (const float*)d_in[9];
  const float* attn_ln_b = (const float*)d_in[10];
  const float* ffn_W1    = (const float*)d_in[11];
  const float* ffn_b1    = (const float*)d_in[12];
  const float* ffn_W2    = (const float*)d_in[13];
  const float* ffn_b2    = (const float*)d_in[14];
  const float* ffn_ln_w  = (const float*)d_in[15];
  const float* ffn_ln_b  = (const float*)d_in[16];
  const float* last_ln_w = (const float*)d_in[17];
  const float* last_ln_b = (const float*)d_in[18];

  char* ws = (char*)d_ws;
  size_t off = 0;
  float* x_work   = (float*)(ws + off); off += (size_t)ROWS * D_MODEL * 4;
  float* ln_buf   = (float*)(ws + off); off += (size_t)ROWS * D_MODEL * 4;
  float* qkv_buf  = (float*)(ws + off); off += (size_t)ROWS * 384 * 4;
  float* attn_o   = (float*)(ws + off); off += (size_t)ROWS * D_ATTN * 4;
  float* posqk    = (float*)(ws + off); off += (size_t)ROWS * 256 * 4;
  float* posattn  = (float*)(ws + off); off += (size_t)SEQ * SEQ * 4;
  float* ffn_chnk = (float*)(ws + off); off += (size_t)SEQ * D_FF * 4;

  hipMemcpyAsync(x_work, x_in, (size_t)ROWS * D_MODEL * 4,
                 hipMemcpyDeviceToDevice, stream);

  // posqk = pos_emb @ pos_qk_W + b
  launch_gemm(stream, pos_emb, D_MODEL, 0, pos_qk_W, 256, 0, 0,
              pos_qk_b, nullptr, posqk, 256, ROWS, 256, D_MODEL, 1, 0);
  // pos_attn[n,m] = sum_b pq[b,n,:] . pk[b,m,:]   (batched, transposed B)
  launch_gemm(stream, posqk, 256, (long long)SEQ * 256,
              posqk + D_ATTN, 256, (long long)SEQ * 256, 1,
              nullptr, nullptr, posattn, SEQ, SEQ, SEQ, D_ATTN, NB, 0);

  const unsigned lnBlocks = ROWS / 8;
  for (int l = 0; l < NLAYER; ++l) {
    ln_kernel<<<lnBlocks, 256, 0, stream>>>(x_work, attn_ln_w + l * D_MODEL,
        attn_ln_b + l * D_MODEL, x_work, 1e-6f);
    launch_gemm(stream, x_work, D_MODEL, 0,
                qkv_W + (size_t)l * D_MODEL * 384, 384, 0, 0,
                qkv_b + (size_t)l * 384, nullptr,
                qkv_buf, 384, ROWS, 384, D_MODEL, 1, 0);
    flash_attn<<<dim3(SEQ / 64, NB), 128, 0, stream>>>(qkv_buf, posattn,
                                                       key_mask, attn_o, SEQ);
    launch_gemm(stream, attn_o, D_ATTN, 0,
                out_W + (size_t)l * D_ATTN * D_MODEL, D_MODEL, 0, 0,
                out_b + (size_t)l * D_MODEL, x_work,
                x_work, D_MODEL, ROWS, D_MODEL, D_ATTN, 1, 0);
    ln_kernel<<<lnBlocks, 256, 0, stream>>>(x_work, ffn_ln_w + l * D_MODEL,
        ffn_ln_b + l * D_MODEL, ln_buf, 1e-8f);
    for (int c = 0; c < 8; ++c) {
      const float* Ac = ln_buf + (size_t)c * SEQ * D_MODEL;
      float*       Xc = x_work + (size_t)c * SEQ * D_MODEL;
      launch_gemm(stream, Ac, D_MODEL, 0,
                  ffn_W1 + (size_t)l * D_MODEL * D_FF, D_FF, 0, 0,
                  ffn_b1 + (size_t)l * D_FF, nullptr,
                  ffn_chnk, D_FF, SEQ, D_FF, D_MODEL, 1, 1);
      launch_gemm(stream, ffn_chnk, D_FF, 0,
                  ffn_W2 + (size_t)l * D_FF * D_MODEL, D_MODEL, 0, 0,
                  ffn_b2 + (size_t)l * D_MODEL, Xc,
                  Xc, D_MODEL, SEQ, D_MODEL, D_FF, 1, 0);
    }
  }
  ln_kernel<<<lnBlocks, 256, 0, stream>>>(x_work, last_ln_w, last_ln_b,
                                          (float*)d_out, 1e-6f);
}
